// EvaAttention_88038239634247
// MI455X (gfx1250) — matrix operand order repaired and checked by reading, not compiled, not run on hardware
//
#include <hip/hip_runtime.h>

// ---------------------------------------------------------------------------
// EVA attention forward for MI455X (gfx1250), bf16 WMMA path, fp32 accum.
// Round 3: fix async global->LDS builtin parameter types (v4i AS1/AS3 ptrs).
// ---------------------------------------------------------------------------

typedef __attribute__((ext_vector_type(8)))  float  v8f;
typedef __attribute__((ext_vector_type(8)))  __bf16 v8bf;
typedef __attribute__((ext_vector_type(16))) __bf16 v16bf;
typedef int v4i __attribute__((vector_size(16)));

constexpr int   kB = 4, kS = 4096, kHID = 1024, kNH = 16, kD = 64;
constexpr int   kWIN = 512, kCHUNK = 128, kCPW = 4, kNW = 8, kNC = 32;
constexpr float kSCALE = 0.125f;   // D^-0.5
constexpr float kNEG   = -1.0e30f;

#if defined(__HIP_DEVICE_COMPILE__) &&                                        \
    __has_builtin(__builtin_amdgcn_global_load_async_to_lds_b128) &&          \
    __has_builtin(__builtin_amdgcn_s_wait_asynccnt)
#define EVA_ASYNC_LDS 1
#else
#define EVA_ASYNC_LDS 0
#endif

// Copy 64 bytes (32 bf16) global -> LDS, per-lane addresses.
static __device__ inline void cp64B(__bf16* dst, const __bf16* src) {
#if EVA_ASYNC_LDS
  v4i* gflat = (v4i*)(void*)const_cast<__bf16*>(src);
  v4i* lflat = (v4i*)(void*)dst;
#pragma unroll
  for (int i = 0; i < 4; ++i) {
    auto g = (__attribute__((address_space(1))) v4i*)(gflat + i);
    auto l = (__attribute__((address_space(3))) v4i*)(lflat + i);
    __builtin_amdgcn_global_load_async_to_lds_b128(g, l, 0, 0);
  }
#else
  const v8bf* s = (const v8bf*)src;
  v8bf* d = (v8bf*)dst;
  d[0] = s[0]; d[1] = s[1]; d[2] = s[2]; d[3] = s[3];
#endif
}
static __device__ inline void async_wait() {
#if EVA_ASYNC_LDS
  __builtin_amdgcn_s_wait_asynccnt(0);
#endif
}

static __device__ inline v16bf cat16(v8bf lo, v8bf hi) {
  v16bf r;
#pragma unroll
  for (int i = 0; i < 8; ++i) { r[i] = lo[i]; r[i + 8] = hi[i]; }
  return r;
}

static __device__ inline float rmax16(float v) {
  v = fmaxf(v, __shfl_xor(v, 1, 32));
  v = fmaxf(v, __shfl_xor(v, 2, 32));
  v = fmaxf(v, __shfl_xor(v, 4, 32));
  v = fmaxf(v, __shfl_xor(v, 8, 32));
  return v;
}
static __device__ inline float rsum16(float v) {
  v += __shfl_xor(v, 1, 32);
  v += __shfl_xor(v, 2, 32);
  v += __shfl_xor(v, 4, 32);
  v += __shfl_xor(v, 8, 32);
  return v;
}

// ---------------------------------------------------------------------------
// fp32 -> bf16 convert (grid-stride)
// ---------------------------------------------------------------------------
__global__ void cvt_f32_to_bf16(const float* __restrict__ in,
                                __bf16* __restrict__ out, int n) {
  for (int i = blockIdx.x * blockDim.x + threadIdx.x; i < n;
       i += gridDim.x * blockDim.x)
    out[i] = (__bf16)in[i];
}

// ---------------------------------------------------------------------------
// C[M,N] = A[M,K](bf16,row-major) * W^T  with W[N,K](bf16,row-major).
// 128x128 C tile per 256-thread block (8 waves).  K staged in 64-wide
// double-buffered LDS tiles filled with async global->LDS copies; each wave
// computes a 32x64 strip = 8 fp32 WMMA accumulators.
// ---------------------------------------------------------------------------
__global__ __launch_bounds__(256) void gemm_bf16_wmma(
    const __bf16* __restrict__ A, const __bf16* __restrict__ W,
    void* __restrict__ Cout, int M, int N, int K, int out_f32) {
  constexpr int GP = 72;  // LDS row pitch (64 data + 8 pad)
  __shared__ alignas(16) __bf16 sA[2][128 * GP];
  __shared__ alignas(16) __bf16 sB[2][128 * GP];

  const int tid  = threadIdx.x;
  const int wave = tid >> 5, lane = tid & 31;
  const int half = (lane >> 4) & 1, n16 = lane & 15;
  const int wm = wave & 3;        // 4 M strips of 32 rows
  const int wn = wave >> 2;       // 2 N strips of 64 cols
  const int tm = blockIdx.x * 128, tn = blockIdx.y * 128;

  // staging role: thread -> (row, 32-col half) of the 128x64 tiles
  const int srow = tid & 127;
  const int sch  = (tid >> 7) & 1;
  const __bf16* gA = A + (size_t)(tm + srow) * (size_t)K + sch * 32;
  const __bf16* gB = W + (size_t)(tn + srow) * (size_t)K + sch * 32;

  auto stage = [&](int b, int kk) {
    cp64B(&sA[b][srow * GP + sch * 32], gA + kk);
    cp64B(&sB[b][srow * GP + sch * 32], gB + kk);
  };

  v8f acc[2][4] = {};

  stage(0, 0);
  async_wait();
  __syncthreads();

  int buf = 0;
  for (int kk = 0; kk < K; kk += 64) {
    if (kk + 64 < K) stage(buf ^ 1, kk + 64);

#pragma unroll
    for (int ks = 0; ks < 2; ++ks) {
      v16bf af[2];
#pragma unroll
      for (int mt = 0; mt < 2; ++mt) {
        const __bf16* ap =
            &sA[buf][(wm * 32 + mt * 16 + n16) * GP + ks * 32 + half * 16];
        af[mt] = cat16(*(const v8bf*)ap, *(const v8bf*)(ap + 8));
      }
#pragma unroll
      for (int nt = 0; nt < 4; ++nt) {
        const __bf16* bp =
            &sB[buf][(wn * 64 + nt * 16 + n16) * GP + ks * 32 + half * 16];
        v16bf bfrag = cat16(*(const v8bf*)bp, *(const v8bf*)(bp + 8));
#pragma unroll
        for (int mt = 0; mt < 2; ++mt)
          acc[mt][nt] = __builtin_amdgcn_wmma_f32_16x16x32_bf16(
              false, af[mt], false, bfrag, (short)0, acc[mt][nt], false, false);
      }
    }
    async_wait();
    __syncthreads();
    buf ^= 1;
  }

#pragma unroll
  for (int mt = 0; mt < 2; ++mt)
#pragma unroll
    for (int nt = 0; nt < 4; ++nt)
#pragma unroll
      for (int r = 0; r < 8; ++r) {
        const size_t m = (size_t)(tm + wm * 32 + mt * 16 + r + 8 * half);
        const size_t n = (size_t)(tn + wn * 64 + nt * 16 + n16);
        const float v = acc[mt][nt][r];
        if (out_f32) ((float*)Cout)[m * N + n] = v;
        else         ((__bf16*)Cout)[m * N + n] = (__bf16)v;
      }
}

// ---------------------------------------------------------------------------
// RoPE in-place on q and k (bf16), one thread per (row, head, d<32) pair.
// ---------------------------------------------------------------------------
__global__ void rope_kernel(__bf16* __restrict__ q, __bf16* __restrict__ k,
                            const float* __restrict__ cosT,
                            const float* __restrict__ sinT, int npairs) {
  const int p = blockIdx.x * blockDim.x + threadIdx.x;
  if (p >= npairs) return;
  const int d = p & 31;
  const int h = (p >> 5) & (kNH - 1);
  const int row = p >> 9;
  const int s = row & (kS - 1);
  const size_t base = (size_t)row * kHID + h * kD;
  const float c1 = cosT[s * kD + d],      s1 = sinT[s * kD + d];
  const float c2 = cosT[s * kD + d + 32], s2 = sinT[s * kD + d + 32];
  {
    float x1 = (float)q[base + d], x2 = (float)q[base + d + 32];
    q[base + d]      = (__bf16)(x1 * c1 - x2 * s1);
    q[base + d + 32] = (__bf16)(x2 * c2 + x1 * s2);
  }
  {
    float x1 = (float)k[base + d], x2 = (float)k[base + d + 32];
    k[base + d]      = (__bf16)(x1 * c1 - x2 * s1);
    k[base + d + 32] = (__bf16)(x2 * c2 + x1 * s2);
  }
}

// ---------------------------------------------------------------------------
// Per-chunk control-variate summaries (beta/gamma softmax, rfa_k/rfa_v).
// ---------------------------------------------------------------------------
__global__ __launch_bounds__(128) void chunk_summary_kernel(
    const __bf16* __restrict__ kb, const __bf16* __restrict__ vb,
    const float* __restrict__ mu, const float* __restrict__ phi,
    __bf16* __restrict__ rfa_k, __bf16* __restrict__ rfa_v) {
  __shared__ float red[128];
  __shared__ float sbeta[128];
  __shared__ float sgamma[128];

  const int bid = blockIdx.x;
  const int c = bid & (kNC - 1);
  const int h = (bid >> 5) & (kNH - 1);
  const int b = bid >> 9;
  const int j = threadIdx.x;

  const size_t krow = ((size_t)(b * kS + c * kCHUNK + j)) * kHID + h * kD;
  float dmu = 0.f, dphi = 0.f, nrm = 0.f;
  for (int d = 0; d < kD; ++d) {
    const float kv = (float)kb[krow + d];
    dmu  += kv * mu[h * kD + d];
    dphi += kv * phi[h * kD + d];
    nrm  += kv * kv;
  }
  const float bl = kSCALE * dmu  - 0.5f * kSCALE * nrm;
  const float gl = kSCALE * dphi - 0.5f * kSCALE * nrm;

  red[j] = bl; __syncthreads();
  for (int s = 64; s > 0; s >>= 1) {
    if (j < s) red[j] = fmaxf(red[j], red[j + s]);
    __syncthreads();
  }
  const float mb = red[0]; __syncthreads();
  const float eb = __expf(bl - mb);
  red[j] = eb; __syncthreads();
  for (int s = 64; s > 0; s >>= 1) {
    if (j < s) red[j] += red[j + s];
    __syncthreads();
  }
  const float sb = red[0]; __syncthreads();
  sbeta[j] = eb / sb;

  red[j] = gl; __syncthreads();
  for (int s = 64; s > 0; s >>= 1) {
    if (j < s) red[j] = fmaxf(red[j], red[j + s]);
    __syncthreads();
  }
  const float mg = red[0]; __syncthreads();
  const float eg = __expf(gl - mg);
  red[j] = eg; __syncthreads();
  for (int s = 64; s > 0; s >>= 1) {
    if (j < s) red[j] += red[j + s];
    __syncthreads();
  }
  const float sg = red[0]; __syncthreads();
  sgamma[j] = eg / sg;
  __syncthreads();

  const size_t obase = (((size_t)(b * kNH + h)) * kNC + c) * kD;
  const int t = threadIdx.x;
  if (t < 64) {
    float acc = 0.f;
    for (int jj = 0; jj < kCHUNK; ++jj)
      acc += sbeta[jj] *
             (float)kb[((size_t)(b * kS + c * kCHUNK + jj)) * kHID + h * kD + t];
    rfa_k[obase + t] = (__bf16)acc;
  } else {
    const int d = t - 64;
    float acc = 0.f;
    for (int jj = 0; jj < kCHUNK; ++jj)
      acc += sgamma[jj] *
             (float)vb[((size_t)(b * kS + c * kCHUNK + jj)) * kHID + h * kD + d];
    rfa_v[obase + d] = (__bf16)acc;
  }
}

// ---------------------------------------------------------------------------
// Flash-style EVA attention (window-causal + chunk-RFA joint softmax).
// ---------------------------------------------------------------------------
__global__ __launch_bounds__(128) void eva_attn_kernel(
    const __bf16* __restrict__ qb, const __bf16* __restrict__ kb,
    const __bf16* __restrict__ vb, const __bf16* __restrict__ rk,
    const __bf16* __restrict__ rv, __bf16* __restrict__ ob) {
  constexpr int Kp = 72, Vp = 136, Pp = 136, Rp = 72, RVp = 40;
  __shared__ alignas(16) __bf16 sK[kCHUNK * Kp];   // K chunk  [j][d]
  __shared__ alignas(16) __bf16 sVT[kD * Vp];      // V chunk^T [d][j]
  __shared__ alignas(16) __bf16 sP[4 * 16 * Pp];   // per-wave P tiles [m][j]
  __shared__ alignas(16) __bf16 sRK[kNC * Rp];     // rfa_k [c][d]
  __shared__ alignas(16) __bf16 sRVT[kD * RVp];    // rfa_v^T [d][c]

  const int tid  = threadIdx.x;
  const int wave = tid >> 5, lane = tid & 31;
  const int half = (lane >> 4) & 1, n16 = lane & 15;
  const int bidx = blockIdx.x;
  const int qblk = bidx & 7;
  const int w    = (bidx >> 3) & 7;
  const int h    = (bidx >> 6) & 15;
  const int b    = bidx >> 10;
  const int qb0  = qblk * 64;
  const size_t win_row0 = (size_t)(b * kS + w * kWIN);

  v16bf aQ[2];
  {
    const size_t rq =
        (win_row0 + qb0 + wave * 16 + n16) * (size_t)kHID + h * kD;
#pragma unroll
    for (int ds = 0; ds < 2; ++ds) {
      const __bf16* p = qb + rq + ds * 32 + half * 16;
      aQ[ds] = cat16(*(const v8bf*)p, *(const v8bf*)(p + 8));
    }
  }

  float m_run[8], l_run[8];
  v8f o[4] = {};
#pragma unroll
  for (int r = 0; r < 8; ++r) { m_run[r] = kNEG; l_run[r] = 0.f; }

  // ---------------- windowed causal part ----------------
  const int lastck = (qb0 + 63) >> 7;
  for (int ck = 0; ck <= lastck; ++ck) {
    __syncthreads();
    {
      const int j = tid;
      const size_t gr =
          (win_row0 + ck * kCHUNK + j) * (size_t)kHID + h * kD;
      cp64B(&sK[j * Kp],      kb + gr);
      cp64B(&sK[j * Kp + 32], kb + gr + 32);
      const __bf16* vs = vb + gr;
#pragma unroll
      for (int d = 0; d < kD; ++d) sVT[d * Vp + j] = vs[d];
    }
    async_wait();
    __syncthreads();

    const bool needmask = (ck * kCHUNK + kCHUNK - 1) > qb0;
    v8f sc[8];
    float mnew[8];
#pragma unroll
    for (int r = 0; r < 8; ++r) mnew[r] = m_run[r];

#pragma unroll
    for (int nt = 0; nt < 8; ++nt) {
      v8f s = {};
#pragma unroll
      for (int ds = 0; ds < 2; ++ds) {
        const __bf16* bp = sK + (nt * 16 + n16) * Kp + ds * 32 + half * 16;
        v16bf bfrag = cat16(*(const v8bf*)bp, *(const v8bf*)(bp + 8));
        s = __builtin_amdgcn_wmma_f32_16x16x32_bf16(
            false, aQ[ds], false, bfrag, (short)0, s, false, false);
      }
      const int jg = ck * kCHUNK + nt * 16 + n16;
#pragma unroll
      for (int r = 0; r < 8; ++r) {
        float v = s[r] * kSCALE;
        const int qi = qb0 + wave * 16 + r + 8 * half;
        if (needmask && jg > qi) v = kNEG;
        s[r] = v;
        mnew[r] = fmaxf(mnew[r], rmax16(v));
      }
      sc[nt] = s;
    }

    float corr[8];
#pragma unroll
    for (int r = 0; r < 8; ++r) {
      corr[r] = __expf(m_run[r] - mnew[r]);
      m_run[r] = mnew[r];
      l_run[r] *= corr[r];
    }
#pragma unroll
    for (int dt = 0; dt < 4; ++dt)
#pragma unroll
      for (int r = 0; r < 8; ++r) o[dt][r] = o[dt][r] * corr[r];

#pragma unroll
    for (int nt = 0; nt < 8; ++nt) {
      v8f s = sc[nt];
#pragma unroll
      for (int r = 0; r < 8; ++r) {
        const float p = __expf(s[r] - m_run[r]);
        l_run[r] += rsum16(p);
        sP[(wave * 16 + r + 8 * half) * Pp + nt * 16 + n16] = (__bf16)p;
      }
    }

#pragma unroll
    for (int dt = 0; dt < 4; ++dt) {
#pragma unroll
      for (int js = 0; js < 4; ++js) {
        const __bf16* ap = sP + (wave * 16 + n16) * Pp + js * 32 + half * 16;
        v16bf af = cat16(*(const v8bf*)ap, *(const v8bf*)(ap + 8));
        const __bf16* bp = sVT + (dt * 16 + n16) * Vp + js * 32 + half * 16;
        v16bf bf = cat16(*(const v8bf*)bp, *(const v8bf*)(bp + 8));
        o[dt] = __builtin_amdgcn_wmma_f32_16x16x32_bf16(
            false, af, false, bf, (short)0, o[dt], false, false);
      }
    }
  }

  // ---------------- chunk-RFA part (strictly earlier windows) ----------------
  const int cvalid = w * kCPW;
  if (cvalid > 0) {
    __syncthreads();
    if (tid < kNC) {
      const int c = tid;
      const size_t gr = (((size_t)(b * kNH + h)) * kNC + c) * kD;
      cp64B(&sRK[c * Rp],      rk + gr);
      cp64B(&sRK[c * Rp + 32], rk + gr + 32);
#pragma unroll
      for (int d = 0; d < kD; ++d) sRVT[d * RVp + c] = rv[gr + d];
    }
    async_wait();
    __syncthreads();

    v8f sc2[2];
    float mnew[8];
#pragma unroll
    for (int r = 0; r < 8; ++r) mnew[r] = m_run[r];

#pragma unroll
    for (int nt = 0; nt < 2; ++nt) {
      v8f s = {};
#pragma unroll
      for (int ds = 0; ds < 2; ++ds) {
        const __bf16* bp = sRK + (nt * 16 + n16) * Rp + ds * 32 + half * 16;
        v16bf bfrag = cat16(*(const v8bf*)bp, *(const v8bf*)(bp + 8));
        s = __builtin_amdgcn_wmma_f32_16x16x32_bf16(
            false, aQ[ds], false, bfrag, (short)0, s, false, false);
      }
      const int c = nt * 16 + n16;
#pragma unroll
      for (int r = 0; r < 8; ++r) {
        float v = s[r] * kSCALE;
        if (c >= cvalid) v = kNEG;
        s[r] = v;
        mnew[r] = fmaxf(mnew[r], rmax16(v));
      }
      sc2[nt] = s;
    }

    float corr[8];
#pragma unroll
    for (int r = 0; r < 8; ++r) {
      corr[r] = __expf(m_run[r] - mnew[r]);
      m_run[r] = mnew[r];
      l_run[r] *= corr[r];
    }
#pragma unroll
    for (int dt = 0; dt < 4; ++dt)
#pragma unroll
      for (int r = 0; r < 8; ++r) o[dt][r] = o[dt][r] * corr[r];

#pragma unroll
    for (int nt = 0; nt < 2; ++nt) {
      v8f s = sc2[nt];
#pragma unroll
      for (int r = 0; r < 8; ++r) {
        const float p = __expf(s[r] - m_run[r]);
        l_run[r] += rsum16(p);
        sP[(wave * 16 + r + 8 * half) * Pp + nt * 16 + n16] = (__bf16)p;
      }
    }

#pragma unroll
    for (int dt = 0; dt < 4; ++dt) {
      const __bf16* ap = sP + (wave * 16 + n16) * Pp + half * 16;
      v16bf af = cat16(*(const v8bf*)ap, *(const v8bf*)(ap + 8));
      const __bf16* bp = sRVT + (dt * 16 + n16) * RVp + half * 16;
      v16bf bf = cat16(*(const v8bf*)bp, *(const v8bf*)(bp + 8));
      o[dt] = __builtin_amdgcn_wmma_f32_16x16x32_bf16(
          false, af, false, bf, (short)0, o[dt], false, false);
    }
  }

  // ---------------- finalize ----------------
#pragma unroll
  for (int dt = 0; dt < 4; ++dt)
#pragma unroll
    for (int r = 0; r < 8; ++r) {
      const float v = o[dt][r] / l_run[r];
      const size_t m = win_row0 + qb0 + wave * 16 + r + 8 * half;
      ob[m * kHID + h * kD + dt * 16 + n16] = (__bf16)v;
    }
}

// ---------------------------------------------------------------------------
// Host orchestration
// ---------------------------------------------------------------------------
extern "C" void kernel_launch(void* const* d_in, const int* in_sizes, int n_in,
                              void* d_out, int out_size, void* d_ws,
                              size_t ws_size, hipStream_t stream) {
  (void)in_sizes; (void)n_in; (void)out_size; (void)ws_size;

  const float* hs   = (const float*)d_in[0];
  const float* Wq   = (const float*)d_in[1];
  const float* Wk   = (const float*)d_in[2];
  const float* Wv   = (const float*)d_in[3];
  const float* Wo   = (const float*)d_in[4];
  const float* mu   = (const float*)d_in[5];
  const float* phi  = (const float*)d_in[6];
  const float* cosT = (const float*)d_in[7];
  const float* sinT = (const float*)d_in[8];

  const size_t HS  = (size_t)kB * kS * kHID;
  const size_t WE  = (size_t)kHID * kHID;
  const size_t RFA = (size_t)kB * kNH * kNC * kD;

  char* ws = (char*)d_ws;
  size_t off = 0;
  auto alloc = [&](size_t bytes) {
    void* p = ws + off;
    off += (bytes + 255) & ~(size_t)255;
    return p;
  };
  __bf16* hs_bf   = (__bf16*)alloc(HS * 2);
  __bf16* wq_bf   = (__bf16*)alloc(WE * 2);
  __bf16* wk_bf   = (__bf16*)alloc(WE * 2);
  __bf16* wv_bf   = (__bf16*)alloc(WE * 2);
  __bf16* wo_bf   = (__bf16*)alloc(WE * 2);
  __bf16* q_bf    = (__bf16*)alloc(HS * 2);
  __bf16* k_bf    = (__bf16*)alloc(HS * 2);
  __bf16* v_bf    = (__bf16*)alloc(HS * 2);
  __bf16* attn_bf = (__bf16*)alloc(HS * 2);
  __bf16* rfa_k   = (__bf16*)alloc(RFA * 2);
  __bf16* rfa_v   = (__bf16*)alloc(RFA * 2);

  cvt_f32_to_bf16<<<2048, 256, 0, stream>>>(hs, hs_bf, (int)HS);
  cvt_f32_to_bf16<<<512, 256, 0, stream>>>(Wq, wq_bf, (int)WE);
  cvt_f32_to_bf16<<<512, 256, 0, stream>>>(Wk, wk_bf, (int)WE);
  cvt_f32_to_bf16<<<512, 256, 0, stream>>>(Wv, wv_bf, (int)WE);
  cvt_f32_to_bf16<<<512, 256, 0, stream>>>(Wo, wo_bf, (int)WE);

  dim3 gg(kB * kS / 128, kHID / 128);
  gemm_bf16_wmma<<<gg, 256, 0, stream>>>(hs_bf, wq_bf, q_bf,
                                         kB * kS, kHID, kHID, 0);
  gemm_bf16_wmma<<<gg, 256, 0, stream>>>(hs_bf, wk_bf, k_bf,
                                         kB * kS, kHID, kHID, 0);
  gemm_bf16_wmma<<<gg, 256, 0, stream>>>(hs_bf, wv_bf, v_bf,
                                         kB * kS, kHID, kHID, 0);

  const int npairs = kB * kS * kNH * 32;
  rope_kernel<<<npairs / 256, 256, 0, stream>>>(q_bf, k_bf, cosT, sinT,
                                                npairs);

  chunk_summary_kernel<<<kB * kNH * kNC, 128, 0, stream>>>(
      k_bf, v_bf, mu, phi, rfa_k, rfa_v);

  eva_attn_kernel<<<kB * kNH * kNW * (kWIN / 64), 128, 0, stream>>>(
      q_bf, k_bf, v_bf, rfa_k, rfa_v, attn_bf);

  gemm_bf16_wmma<<<gg, 256, 0, stream>>>(attn_bf, wo_bf, d_out,
                                         kB * kS, kHID, kHID, 1);
}
